// MultiHeadAttention_4999341933079
// MI455X (gfx1250) — matrix operand order repaired and checked by reading, not compiled, not run on hardware
//
#include <hip/hip_runtime.h>
#include <hip/hip_bf16.h>
#include <stdint.h>

// ---------------------------------------------------------------- types
typedef __bf16 bf16;
typedef __attribute__((ext_vector_type(16))) __bf16 v16bf;
typedef __attribute__((ext_vector_type(8)))  __bf16 v8bf;
typedef __attribute__((ext_vector_type(8)))  float  v8f;
typedef __attribute__((ext_vector_type(4)))  unsigned int v4u;
typedef __attribute__((ext_vector_type(8)))  int v8i32;
typedef __attribute__((ext_vector_type(4)))  int v4i32;

#define DIM    768
#define HEADS  16
#define DH     48
#define SEQ    1024
#define BATCH  8
#define NROWS  (BATCH*SEQ)     // 8192
#define NQKV   (3*DIM)         // 2304
#define BH     (BATCH*HEADS)   // 128
#define HSZ    (SEQ*DH)        // 49152 elements per (b,h)
#define NKSTEP (DIM/32)        // 24 k-steps
#define ATTN_SCALE 0.14433756729740643f  // 48^-0.5

union ABu { v16bf v; v8bf h[2]; unsigned u[8]; };

__device__ __forceinline__ v8f zero8f() {
  v8f z;
#pragma unroll
  for (int i = 0; i < 8; i++) z[i] = 0.0f;
  return z;
}

__device__ __forceinline__ v8f wmma_bf16(v16bf a, v16bf b, v8f c) {
  // D = A(16x32 bf16) * B(32x16 bf16) + C(16x16 f32)
  return __builtin_amdgcn_wmma_f32_16x16x32_bf16(false, a, false, b, (short)0, c,
                                                 false, false);
}

// ------------------------------------------------- WMMA fragment loaders
// A-matrix 16x32 (MxK): lanes 0-15 M=lane, K blocks {0-7,16-23}; lanes 16-31
// same rows, K +8.  LDS tile M-major, K contiguous.
__device__ __forceinline__ v16bf ldsA_frag(const bf16* ls, int ld, int row0) {
  const int lane = threadIdx.x & 31;
  const int m = lane & 15, hf = lane >> 4;
  const bf16* p = ls + (size_t)(row0 + m) * ld + hf * 8;
  ABu r;
  r.h[0] = *(const v8bf*)(p);
  r.h[1] = *(const v8bf*)(p + 16);
  return r.v;
}

// B-matrix 32x16 (KxN): lane gives N, K blocks {0-7,16-23}; lanes 16-31 K +8.
// LDS tile N-major, K contiguous.
__device__ __forceinline__ v16bf ldsB_frag(const bf16* ls, int ld, int col0) {
  const int lane = threadIdx.x & 31;
  const int n = lane & 15, hf = lane >> 4;
  const bf16* p = ls + (size_t)(col0 + n) * ld + hf * 8;
  ABu r;
  r.h[0] = *(const v8bf*)(p);
  r.h[1] = *(const v8bf*)(p + 16);
  return r.v;
}

// --------------------------------------------------------- TDM (gfx1250)
#if defined(__has_builtin)
#if __has_builtin(__builtin_amdgcn_tensor_load_to_lds) && \
    __has_builtin(__builtin_amdgcn_s_wait_tensorcnt)
#define USE_TDM 1
#endif
#endif
#ifndef USE_TDM
#define USE_TDM 0
#endif

#if USE_TDM
// 2D tile load: tile_d0 elements contiguous per row, tile_d1 rows with
// stride0 (elements), data_size = 2 bytes (bf16).
__device__ __forceinline__ void tdm_load_2d(unsigned lds_off, const bf16* gaddr,
                                            unsigned tensor_d0, unsigned tensor_d1,
                                            unsigned stride0,
                                            unsigned tile_d0, unsigned tile_d1) {
  unsigned long long ga = (unsigned long long)(uintptr_t)gaddr;
  v4u g0;
  g0.x = 1u;                                      // count=1, user descriptor
  g0.y = lds_off;                                 // lds_addr (bytes)
  g0.z = (unsigned)(ga & 0xFFFFFFFFu);            // global_addr[31:0]
  g0.w = (unsigned)((ga >> 32) & 0x01FFFFFFu) | (2u << 30);  // addr[56:32]|type=2
  v8i32 g1;
  g1[0] = (int)(1u << 16);                        // data_size=1 -> 2 bytes
  g1[1] = (int)((tensor_d0 & 0xFFFFu) << 16);     // tensor_dim0[15:0]
  g1[2] = (int)((tensor_d0 >> 16) | ((tensor_d1 & 0xFFFFu) << 16));
  g1[3] = (int)((tensor_d1 >> 16) | (tile_d0 << 16));   // tile_dim0
  g1[4] = (int)(tile_d1 & 0xFFFFu);               // tile_dim1 (tile_dim2=0)
  g1[5] = (int)stride0;                           // tensor_dim0_stride[31:0]
  g1[6] = 0;
  g1[7] = 0;
  v4i32 z4 = {0, 0, 0, 0};
  v8i32 z8;
#pragma unroll
  for (int i = 0; i < 8; i++) z8[i] = 0;
  __builtin_amdgcn_tensor_load_to_lds(g0, g1, z4, z4, z8, 0);
}
#endif

// ----------------------------------------------------------- GEMM core
// 256x128 block tile, 256 threads (8 waves as 4(M) x 2(N)); each wave owns a
// 64x64 tile = 4x4 WMMA fragments (16 ds_load_b128 -> 16 WMMA per k-step).
// Software-pipelined with double-buffered LDS: next tile's global loads are
// issued before compute, ds_stores land after compute.
__device__ __forceinline__ void gemm_mainloop(const bf16* __restrict__ A,
                                              const bf16* __restrict__ BT,
                                              int rowBase, int colBase,
                                              bf16* lsA, bf16* lsB,
                                              v8f acc[4][4]) {
  const int t = threadIdx.x;
  const int w = t >> 5;
  const int wm = w & 3, wn = w >> 2;
  const bf16* ga = A + (size_t)(rowBase + t) * DIM;        // 1 A row / thread
  const int sb = t >> 1, sc = (t & 1) * 16;
  const bf16* gb = BT + (size_t)(colBase + sb) * DIM + sc; // 16 B elems / thr
  const int soA = t * 32;
  const int soB = sb * 32 + sc;

  // prologue: stage k-step 0 into buffer 0
  {
    v8bf a0 = *(const v8bf*)(ga),      a1 = *(const v8bf*)(ga + 8);
    v8bf a2 = *(const v8bf*)(ga + 16), a3 = *(const v8bf*)(ga + 24);
    v8bf b0 = *(const v8bf*)(gb),      b1 = *(const v8bf*)(gb + 8);
    *(v8bf*)(lsA + soA)      = a0; *(v8bf*)(lsA + soA + 8)  = a1;
    *(v8bf*)(lsA + soA + 16) = a2; *(v8bf*)(lsA + soA + 24) = a3;
    *(v8bf*)(lsB + soB)      = b0; *(v8bf*)(lsB + soB + 8)  = b1;
  }

  for (int k = 0; k < NKSTEP; k++) {
    const int cur = k & 1;
    __syncthreads();                      // staged data visible; nxt buf free
    v8bf na0, na1, na2, na3, nb0, nb1;
    const bool has_next = (k + 1 < NKSTEP);
    if (has_next) {                       // issue next tile's global loads
      const bf16* pa = ga + (k + 1) * 32;
      const bf16* pb = gb + (k + 1) * 32;
      na0 = *(const v8bf*)(pa);      na1 = *(const v8bf*)(pa + 8);
      na2 = *(const v8bf*)(pa + 16); na3 = *(const v8bf*)(pa + 24);
      nb0 = *(const v8bf*)(pb);      nb1 = *(const v8bf*)(pb + 8);
    }
    const bf16* cA = lsA + cur * (256 * 32);
    const bf16* cB = lsB + cur * (128 * 32);
    v16bf af[4], bfr[4];
#pragma unroll
    for (int fm = 0; fm < 4; fm++) af[fm] = ldsA_frag(cA, 32, wm * 64 + fm * 16);
#pragma unroll
    for (int fn = 0; fn < 4; fn++) bfr[fn] = ldsB_frag(cB, 32, wn * 64 + fn * 16);
#pragma unroll
    for (int fm = 0; fm < 4; fm++)
#pragma unroll
      for (int fn = 0; fn < 4; fn++)
        acc[fm][fn] = wmma_bf16(af[fm], bfr[fn], acc[fm][fn]);
    if (has_next) {                       // park next tile in other buffer
      bf16* nA = lsA + (cur ^ 1) * (256 * 32);
      bf16* nB = lsB + (cur ^ 1) * (128 * 32);
      *(v8bf*)(nA + soA)      = na0; *(v8bf*)(nA + soA + 8)  = na1;
      *(v8bf*)(nA + soA + 16) = na2; *(v8bf*)(nA + soA + 24) = na3;
      *(v8bf*)(nB + soB)      = nb0; *(v8bf*)(nB + soB + 8)  = nb1;
    }
  }
}

// ------------------------------------------------------- convert kernels
__global__ void k_cvt8(const float* __restrict__ in, bf16* __restrict__ out,
                       int n) {
  int i = (blockIdx.x * blockDim.x + threadIdx.x) * 8;
  if (i < n) {
    float4 f0 = *(const float4*)(in + i);
    float4 f1 = *(const float4*)(in + i + 4);
    v8bf o;
    o[0] = (bf16)f0.x; o[1] = (bf16)f0.y; o[2] = (bf16)f0.z; o[3] = (bf16)f0.w;
    o[4] = (bf16)f1.x; o[5] = (bf16)f1.y; o[6] = (bf16)f1.z; o[7] = (bf16)f1.w;
    *(v8bf*)(out + i) = o;
  }
}

// in: [K][N] f32 row-major -> outT: [N][K] bf16, 32x32 LDS-tiled transpose.
__global__ __launch_bounds__(256) void k_cvt_T(const float* __restrict__ in,
                                               bf16* __restrict__ outT,
                                               int K, int N) {
  __shared__ bf16 tile[32][33];
  const int kb = blockIdx.x * 32, nb = blockIdx.y * 32;
  const int tx = threadIdx.x & 31, ty = threadIdx.x >> 5;  // 8 row-groups
#pragma unroll
  for (int yy = ty; yy < 32; yy += 8)
    tile[yy][tx] = (bf16)in[(size_t)(kb + yy) * N + nb + tx];  // coalesced (N)
  __syncthreads();
#pragma unroll
  for (int yy = ty; yy < 32; yy += 8)
    outT[(size_t)(nb + yy) * K + kb + tx] = tile[tx][yy];      // coalesced (K)
}

// -------------------------------------------------------- QKV projection
// qkv = x @ Wqkv ; epilogue scatters through the faithful torch view
// (3,B,HEADS,SEQ,DH) into Q[bh][n][d], K[bh][n][d], V^T[bh][d][n] (bf16).
// Index math: flat = R*2304 + C; with t = 3R + C/768:
//   sel = t>>13, bh = (t&8191)>>6, nn = (t&63)*16 + (C%768)/48, dd = (C%768)%48
__global__ __launch_bounds__(256) void k_gemm_qkv(const bf16* __restrict__ xb,
                                                  const bf16* __restrict__ wqkvT,
                                                  bf16* __restrict__ Qb,
                                                  bf16* __restrict__ Kb,
                                                  bf16* __restrict__ VTb) {
  __shared__ __align__(16) bf16 lsA[2 * 256 * 32];
  __shared__ __align__(16) bf16 lsB[2 * 128 * 32];
  v8f acc[4][4];
#pragma unroll
  for (int fm = 0; fm < 4; fm++)
#pragma unroll
    for (int fn = 0; fn < 4; fn++) acc[fm][fn] = zero8f();

  const int rowBase = blockIdx.x * 256, colBase = blockIdx.y * 128;
  gemm_mainloop(xb, wqkvT, rowBase, colBase, lsA, lsB, acc);

  const int lane = threadIdx.x & 31, w = threadIdx.x >> 5;
  const int wm = w & 3, wn = w >> 2;
  const int ln = lane & 15, hf = lane >> 4;
#pragma unroll
  for (int fm = 0; fm < 4; fm++)
#pragma unroll
    for (int fn = 0; fn < 4; fn++) {
      const int C = colBase + wn * 64 + fn * 16 + ln;
      const int c_hi = C / 768;              // 0..2
      const int c_lo = C - c_hi * 768;
      const int nn_lo = c_lo / 48;           // 0..15
      const int dd = c_lo - nn_lo * 48;      // 0..47
      const int Rb = rowBase + wm * 64 + fm * 16 + hf * 8;
      const int tqb = 3 * Rb + c_hi;
#pragma unroll
      for (int v = 0; v < 8; v++) {
        const int tq = tqb + 3 * v;
        const int sel = tq >> 13;
        const int tm = tq & 8191;
        const int bh = tm >> 6;
        const int nn = (tm & 63) * 16 + nn_lo;
        bf16 val = (bf16)acc[fm][fn][v];
        if (sel == 0)      Qb[(size_t)bh * HSZ + (size_t)nn * DH + dd] = val;
        else if (sel == 1) Kb[(size_t)bh * HSZ + (size_t)nn * DH + dd] = val;
        else               VTb[(size_t)bh * HSZ + (size_t)dd * SEQ + nn] = val;
      }
    }
}

// ------------------------------------------------------------ attention
__device__ __forceinline__ float halfmax16(float x) {
#pragma unroll
  for (int m = 1; m <= 8; m <<= 1) x = fmaxf(x, __shfl_xor(x, m, 32));
  return x;
}
__device__ __forceinline__ float halfsum16(float x) {
#pragma unroll
  for (int m = 1; m <= 8; m <<= 1) x += __shfl_xor(x, m, 32);
  return x;
}

// B-frag from K-tile in LDS (N-major keys x 48 d).  kstep1 covers d32..63:
// elements 8-15 (d48..63) are the zero pad.
__device__ __forceinline__ v16bf kfragB(const bf16* ls, int col0, int kstep) {
  const int lane = threadIdx.x & 31;
  const int n = lane & 15, hf = lane >> 4;
  const bf16* p = ls + (size_t)(col0 + n) * DH;
  ABu r;
#pragma unroll
  for (int i = 0; i < 8; i++) r.u[i] = 0;
  if (kstep == 0) {
    r.h[0] = *(const v8bf*)(p + hf * 8);
    r.h[1] = *(const v8bf*)(p + 16 + hf * 8);
  } else {
    r.h[0] = *(const v8bf*)(p + 32 + hf * 8);
  }
  return r.v;
}

// A-frag of softmax probabilities from per-wave LDS strip [16 rows][128 keys]
__device__ __forceinline__ v16bf pfrag(const bf16* ls, int j) {
  const int lane = threadIdx.x & 31;
  const int m = lane & 15, hf = lane >> 4;
  const bf16* p = ls + (size_t)m * 128 + j * 32 + hf * 8;
  ABu r;
  r.h[0] = *(const v8bf*)(p);
  r.h[1] = *(const v8bf*)(p + 16);
  return r.v;
}

// B-frag from V^T tile in LDS ([48 d rows][128 keys], keys contiguous)
__device__ __forceinline__ v16bf vfrag(const bf16* ls, int col0, int j) {
  const int lane = threadIdx.x & 31;
  const int n = lane & 15, hf = lane >> 4;
  const bf16* p = ls + (size_t)(col0 + n) * 128 + j * 32 + hf * 8;
  ABu r;
  r.h[0] = *(const v8bf*)(p);
  r.h[1] = *(const v8bf*)(p + 16);
  return r.v;
}

#define KTILE (128 * DH)   // 6144 elements per K/V tile

__global__ __launch_bounds__(256) void k_attn(const bf16* __restrict__ Qg,
                                              const bf16* __restrict__ Kg,
                                              const bf16* __restrict__ VTg,
                                              bf16* __restrict__ ctx) {
  __shared__ __align__(16) bf16 lsK[2 * KTILE];     // keys x d, double buffer
  __shared__ __align__(16) bf16 lsV[2 * KTILE];     // d x keys (V^T), double
  __shared__ __align__(16) bf16 lsP[8 * 16 * 128];  // per-wave P strips
  const int t = threadIdx.x;
  const int w = t >> 5, lane = t & 31;
  const int ln = lane & 15, hf = lane >> 4;
  const int bh = blockIdx.x;
  const int q0 = blockIdx.y * 128 + w * 16;
  bf16* lsPw = lsP + (size_t)w * 16 * 128;
  const bf16* Kbase = Kg + (size_t)bh * HSZ;
  const bf16* Vbase = VTg + (size_t)bh * HSZ;

  // Q fragments (A-layout), dh padded 48 -> 64 with zero registers
  const bf16* Qrow = Qg + (size_t)bh * HSZ + (size_t)(q0 + ln) * DH;
  ABu qa0, qa1;
  qa0.h[0] = *(const v8bf*)(Qrow + hf * 8);
  qa0.h[1] = *(const v8bf*)(Qrow + 16 + hf * 8);
#pragma unroll
  for (int i = 0; i < 8; i++) qa1.u[i] = 0;
  qa1.h[0] = *(const v8bf*)(Qrow + 32 + hf * 8);   // d32..47 ; hi half stays 0

  float m_i[8], l_i[8];
#pragma unroll
  for (int v = 0; v < 8; v++) { m_i[v] = -1e30f; l_i[v] = 0.0f; }
  v8f o[3];
#pragma unroll
  for (int c = 0; c < 3; c++) o[c] = zero8f();

  const int NT = SEQ / 128;
#if USE_TDM
  if (t < 32) {   // wave 0 drives the TDM, one tile ahead
    tdm_load_2d((unsigned)(uintptr_t)(void*)lsK, Kbase, DH, 128, DH, DH, 128);
    tdm_load_2d((unsigned)(uintptr_t)(void*)lsV, Vbase, SEQ, DH, SEQ, 128, DH);
  }
#endif

  for (int kt = 0; kt < NT; kt++) {
    const int cur = kt & 1;
    const bf16* cK = lsK + cur * KTILE;
    const bf16* cV = lsV + cur * KTILE;
    __syncthreads();   // prev compute done -> other buffer free for TDM
#if USE_TDM
    if (t < 32) {
      if (kt + 1 < NT) {
        unsigned nxt = (unsigned)(cur ^ 1) * (KTILE * 2);  // byte offset delta
        tdm_load_2d((unsigned)(uintptr_t)(void*)lsK + nxt,
                    Kbase + (size_t)(kt + 1) * KTILE, DH, 128, DH, DH, 128);
        tdm_load_2d((unsigned)(uintptr_t)(void*)lsV + nxt,
                    Vbase + (kt + 1) * 128, SEQ, DH, SEQ, 128, DH);
        __builtin_amdgcn_s_wait_tensorcnt(2);  // current tile's pair done
      } else {
        __builtin_amdgcn_s_wait_tensorcnt(0);
      }
    }
#else
    {
      const bf16* gk = Kbase + (size_t)kt * KTILE;
      bf16* dK = (bf16*)cK;
      for (int i = t; i < 768; i += 256)          // 128*48 contiguous
        *(v8bf*)(dK + i * 8) = *(const v8bf*)(gk + i * 8);
      const bf16* gv = Vbase + kt * 128;
      bf16* dV = (bf16*)cV;
      for (int i = t; i < 768; i += 256) {
        int dd = i >> 4, cc = (i & 15) * 8;
        *(v8bf*)(dV + dd * 128 + cc) = *(const v8bf*)(gv + (size_t)dd * SEQ + cc);
      }
    }
#endif
    __syncthreads();

    // S = Q K^T  (16 x 128 per wave)
    v8f s[8];
#pragma unroll
    for (int n = 0; n < 8; n++) {
      s[n] = zero8f();
      s[n] = wmma_bf16(qa0.v, kfragB(cK, n * 16, 0), s[n]);
      s[n] = wmma_bf16(qa1.v, kfragB(cK, n * 16, 1), s[n]);
    }

    // online softmax over this key tile
#pragma unroll
    for (int v = 0; v < 8; v++) {
      float mx = s[0][v];
#pragma unroll
      for (int n = 1; n < 8; n++) mx = fmaxf(mx, s[n][v]);
      mx = halfmax16(mx) * ATTN_SCALE;
      float mnew = fmaxf(m_i[v], mx);
      float alpha = __expf(m_i[v] - mnew);
      float rs = 0.0f;
#pragma unroll
      for (int n = 0; n < 8; n++) {
        float p = __expf(s[n][v] * ATTN_SCALE - mnew);
        rs += p;
        lsPw[(v + hf * 8) * 128 + n * 16 + ln] = (bf16)p;
      }
      rs = halfsum16(rs);
      l_i[v] = l_i[v] * alpha + rs;
      m_i[v] = mnew;
#pragma unroll
      for (int c = 0; c < 3; c++) o[c][v] *= alpha;
    }

    // O += P @ V  (per-wave LDS strip; same-wave DS ordering)
#pragma unroll
    for (int j = 0; j < 4; j++) {
      v16bf pa = pfrag(lsPw, j);
#pragma unroll
      for (int c = 0; c < 3; c++)
        o[c] = wmma_bf16(pa, vfrag(cV, c * 16, j), o[c]);
    }
  }

  // normalize and emit context (bf16, head-major = faithful concat layout)
  bf16* crow = ctx + (size_t)bh * HSZ;
#pragma unroll
  for (int v = 0; v < 8; v++) {
    float inv = 1.0f / l_i[v];
    int row = q0 + v + hf * 8;
#pragma unroll
    for (int c = 0; c < 3; c++)
      crow[(size_t)row * DH + c * 16 + ln] = (bf16)(o[c][v] * inv);
  }
}

// ------------------------------------------------------ output projection
__global__ __launch_bounds__(256) void k_gemm_out(const bf16* __restrict__ ctxb,
                                                  const bf16* __restrict__ woT,
                                                  const float* __restrict__ bias,
                                                  float* __restrict__ out) {
  __shared__ __align__(16) bf16 lsA[2 * 256 * 32];
  __shared__ __align__(16) bf16 lsB[2 * 128 * 32];
  v8f acc[4][4];
#pragma unroll
  for (int fm = 0; fm < 4; fm++)
#pragma unroll
    for (int fn = 0; fn < 4; fn++) acc[fm][fn] = zero8f();

  const int rowBase = blockIdx.x * 256, colBase = blockIdx.y * 128;
  gemm_mainloop(ctxb, woT, rowBase, colBase, lsA, lsB, acc);

  const int lane = threadIdx.x & 31, w = threadIdx.x >> 5;
  const int wm = w & 3, wn = w >> 2;
  const int ln = lane & 15, hf = lane >> 4;
#pragma unroll
  for (int fm = 0; fm < 4; fm++)
#pragma unroll
    for (int fn = 0; fn < 4; fn++) {
      int C = colBase + wn * 64 + fn * 16 + ln;
      float b = bias[C];
#pragma unroll
      for (int v = 0; v < 8; v++) {
        int R = rowBase + wm * 64 + fm * 16 + hf * 8 + v;
        out[(size_t)R * DIM + C] = acc[fm][fn][v] + b;
      }
    }
}

// -------------------------------------------------------------- launcher
extern "C" void kernel_launch(void* const* d_in, const int* in_sizes, int n_in,
                              void* d_out, int out_size, void* d_ws, size_t ws_size,
                              hipStream_t stream) {
  (void)in_sizes; (void)n_in; (void)out_size; (void)ws_size;
  const float* x    = (const float*)d_in[0];
  const float* Wqkv = (const float*)d_in[1];
  const float* Wo   = (const float*)d_in[2];
  const float* bo   = (const float*)d_in[3];
  float* out = (float*)d_out;

  size_t off = 0;
  auto carve = [&](size_t elems) -> bf16* {
    bf16* p = (bf16*)((char*)d_ws + off);
    off += ((elems * sizeof(bf16)) + 255) & ~(size_t)255;
    return p;
  };
  bf16* xb    = carve((size_t)NROWS * DIM);
  bf16* wqkvT = carve((size_t)NQKV * DIM);
  bf16* woT   = carve((size_t)DIM * DIM);
  bf16* Qb    = carve((size_t)BH * HSZ);
  bf16* Kb    = carve((size_t)BH * HSZ);
  bf16* VTb   = carve((size_t)BH * HSZ);
  bf16* ctxb  = carve((size_t)BH * HSZ);

  k_cvt8<<<(NROWS * DIM / 8 + 255) / 256, 256, 0, stream>>>(x, xb, NROWS * DIM);
  k_cvt_T<<<dim3(DIM / 32, NQKV / 32), 256, 0, stream>>>(Wqkv, wqkvT, DIM, NQKV);
  k_cvt_T<<<dim3(DIM / 32, DIM / 32), 256, 0, stream>>>(Wo, woT, DIM, DIM);
  k_gemm_qkv<<<dim3(NROWS / 256, NQKV / 128), 256, 0, stream>>>(xb, wqkvT, Qb, Kb, VTb);
  k_attn<<<dim3(BH, SEQ / 128), 256, 0, stream>>>(Qb, Kb, VTb, ctxb);
  k_gemm_out<<<dim3(NROWS / 256, DIM / 128), 256, 0, stream>>>(ctxb, woT, bo, out);
}
